// AttentionDCA_60309930770664
// MI455X (gfx1250) — matrix operand order repaired and checked by reading, not compiled, not run on hardware
//
#include <hip/hip_runtime.h>
#include <hip/hip_bf16.h>

typedef __attribute__((ext_vector_type(16))) _Float16 v16h;
typedef __attribute__((ext_vector_type(8)))  _Float16 v8h;
typedef __attribute__((ext_vector_type(8)))  float    v8f;

// Problem dims (fixed by reference)
#define Bn 64
#define Ln 512
#define Hn 32
#define Dn 128
#define NAA 21
#define QK_ELEMS (Hn * Ln * Dn)   // 2,097,152 per tensor

// ---------------------------------------------------------------------------
// Kernel 0: one-shot f32 -> f16 conversion of Q and K (removes all cvt VALU
// from the GEMM hot loop; halves fragment load bytes).
// ---------------------------------------------------------------------------
__global__ __launch_bounds__(256)
void dca_cvt_kernel(const float* __restrict__ Q,
                    const float* __restrict__ K,
                    _Float16* __restrict__ Qh,
                    _Float16* __restrict__ Kh) {
    const int i = (blockIdx.x * 256 + threadIdx.x) * 4;
    const float4 q = *(const float4*)(Q + i);
    const float4 k = *(const float4*)(K + i);
    v8h out;
    out[0] = (_Float16)q.x; out[1] = (_Float16)q.y;
    out[2] = (_Float16)q.z; out[3] = (_Float16)q.w;
    *(__attribute__((ext_vector_type(4))) _Float16*)(Qh + i) =
        __builtin_shufflevector(out, out, 0, 1, 2, 3);
    out[0] = (_Float16)k.x; out[1] = (_Float16)k.y;
    out[2] = (_Float16)k.z; out[3] = (_Float16)k.w;
    *(__attribute__((ext_vector_type(4))) _Float16*)(Kh + i) =
        __builtin_shufflevector(out, out, 0, 1, 2, 3);
}

// ---------------------------------------------------------------------------
// Kernel 1: S[h] = Qh[h] (LxD) * Kh[h]^T (DxL) * (1/128) via
// v_wmma_f32_16x16x32_f16. One wave per 16x512 strip, 4 waves / block.
// Fragment layouts per CDNA5 ISA 7.12.2:
//   A 16x32 f16 : lane m = t&15, half = t>>4; elems 0..7 -> K = half*8+e,
//                 elems 8..15 -> K = 16 + half*8 + (e-8)
//   B 32x16 f16 : lane n = t&15, khalf = t>>4; elems 0..7 -> K = khalf*8+e, 8..15 -> 16+khalf*8+(e-8)
//   C/D 16x16 f32: lane n = t&15; VGPR r -> row m = r + 8*(t>>4)
// All fragment chunks are contiguous 8 x f16 = 16B aligned -> b128 loads.
// ---------------------------------------------------------------------------
__global__ __launch_bounds__(128)
void dca_qk_wmma_kernel(const _Float16* __restrict__ Qh,
                        const _Float16* __restrict__ Kh,
                        float* __restrict__ S) {
    const int lane = threadIdx.x & 31;
    const int wave = threadIdx.x >> 5;
    const int strip = blockIdx.x * 4 + wave;        // H * (L/16) strips = 1024
    const int h    = strip >> 5;                    // 32 strips per head
    const int row0 = (strip & 31) << 4;

    const int m    = lane & 15;
    const int half = lane >> 4;
    const int n    = m;

    // ---- A fragments (Q rows), loaded once, reused across all 32 j-tiles
    const _Float16* qrow = Qh + ((size_t)(h * Ln + row0 + m)) * Dn;
    v16h afrag[4];
#pragma unroll
    for (int kb = 0; kb < 4; ++kb) {
        const int base = kb * 32 + half * 8;
        const v8h lo = *(const v8h*)(qrow + base);
        const v8h hi = *(const v8h*)(qrow + base + 16);
        afrag[kb] = __builtin_shufflevector(lo, hi,
            0, 1, 2, 3, 4, 5, 6, 7, 8, 9, 10, 11, 12, 13, 14, 15);
    }

    // ---- Sweep j-tiles
    for (int jt = 0; jt < Ln / 16; ++jt) {
        const int col0 = jt << 4;
        const _Float16* krow = Kh + ((size_t)(h * Ln + col0 + n)) * Dn;
        v8f acc = {};
#pragma unroll
        for (int kb = 0; kb < 4; ++kb) {
            const int base = kb * 32 + half * 8;    // two 8 f16 chunks, K = half*8 and 16+half*8
            const v8h lo = *(const v8h*)(krow + base);
            const v8h hi = *(const v8h*)(krow + base + 16);
            const v16h bfrag = __builtin_shufflevector(lo, hi,
                0, 1, 2, 3, 4, 5, 6, 7, 8, 9, 10, 11, 12, 13, 14, 15);
            acc = __builtin_amdgcn_wmma_f32_16x16x32_f16(
                false, afrag[kb], false, bfrag, (short)0, acc, false, false);
        }
        // ---- Scatter D fragment (scaled by 1/d_k = 1/128)
#pragma unroll
        for (int r = 0; r < 8; ++r) {
            const int mrow = r + 8 * half;
            S[((size_t)(h * Ln + row0 + mrow)) * Ln + col0 + n] = acc[r] * 0.0078125f;
        }
    }
}

// ---------------------------------------------------------------------------
// Kernel 2: in-place row softmax over the last axis (length 512).
// One wave32 per row, 16 elements/lane, shuffle reductions.
// ---------------------------------------------------------------------------
__global__ __launch_bounds__(256)
void dca_softmax_kernel(float* __restrict__ S) {
    const int lane = threadIdx.x & 31;
    const int wave = threadIdx.x >> 5;
    const int row  = blockIdx.x * 8 + wave;         // H*L = 16384 rows
    float* __restrict__ p = S + (size_t)row * Ln;

    float v[16];
#pragma unroll
    for (int k = 0; k < 16; ++k) v[k] = p[lane + 32 * k];

    float mx = v[0];
#pragma unroll
    for (int k = 1; k < 16; ++k) mx = fmaxf(mx, v[k]);
#pragma unroll
    for (int off = 16; off > 0; off >>= 1)
        mx = fmaxf(mx, __shfl_xor(mx, off, 32));

    float sum = 0.f;
#pragma unroll
    for (int k = 0; k < 16; ++k) { v[k] = __expf(v[k] - mx); sum += v[k]; }
#pragma unroll
    for (int off = 16; off > 0; off >>= 1)
        sum += __shfl_xor(sum, off, 32);

    const float inv = 1.0f / sum;
#pragma unroll
    for (int k = 0; k < 16; ++k) p[lane + 32 * k] = v[k] * inv;
}

// ---------------------------------------------------------------------------
// Kernel 3: energy[b] -= sum_{i,j} A[h,i,j] * V[h, x[b,i], x[b,j]]
// One block per (b,h). x_b, V_h in LDS; x[j] register-cached per thread.
// A (33.5 MB) is L2-resident across the 64 batch passes.
// ---------------------------------------------------------------------------
__global__ __launch_bounds__(256)
void dca_energy_kernel(const float* __restrict__ A,
                       const int*   __restrict__ x,
                       const float* __restrict__ V,
                       float* __restrict__ out) {
    __shared__ float Vsh[NAA * NAA];
    __shared__ int   xsh[Ln];
    __shared__ float red[256];

    const int h = blockIdx.x & (Hn - 1);
    const int b = blockIdx.x >> 5;
    const int tid = threadIdx.x;

    if (tid < NAA * NAA) Vsh[tid] = V[h * NAA * NAA + tid];
    for (int i = tid; i < Ln; i += 256) xsh[i] = x[(size_t)b * Ln + i];
    __syncthreads();

    // Each thread owns columns j = tid and j = tid + 256 for every row.
    const int xj0 = xsh[tid];
    const int xj1 = xsh[tid + 256];

    float sum = 0.f;
    const float* __restrict__ Ah = A + (size_t)h * Ln * Ln;
    for (int i = 0; i < Ln; ++i) {
        const int rb = xsh[i] * NAA;                // LDS broadcast read
        const float* __restrict__ Arow = Ah + (size_t)i * Ln;
        sum = fmaf(Arow[tid],       Vsh[rb + xj0], sum);
        sum = fmaf(Arow[tid + 256], Vsh[rb + xj1], sum);
    }

    red[tid] = sum;
    __syncthreads();
#pragma unroll
    for (int s = 128; s > 0; s >>= 1) {
        if (tid < s) red[tid] += red[tid + s];
        __syncthreads();
    }
    if (tid == 0) atomicAdd(&out[b], -red[0]);
}

// ---------------------------------------------------------------------------
extern "C" void kernel_launch(void* const* d_in, const int* in_sizes, int n_in,
                              void* d_out, int out_size, void* d_ws, size_t ws_size,
                              hipStream_t stream) {
    (void)in_sizes; (void)n_in; (void)out_size; (void)ws_size;
    const int*   x = (const int*)  d_in[0];
    const float* Q = (const float*)d_in[1];
    const float* K = (const float*)d_in[2];
    const float* V = (const float*)d_in[3];
    float* out = (float*)d_out;

    // Workspace layout: [Qh f16 4MB][Kh f16 4MB][S f32 33.5MB]
    _Float16* Qh = (_Float16*)d_ws;
    _Float16* Kh = Qh + QK_ELEMS;
    float* Sbuf  = (float*)((char*)d_ws + 2 * QK_ELEMS * sizeof(_Float16));

    // 0) f32 -> f16 pre-conversion of Q and K (4 elems/thread)
    dca_cvt_kernel<<<QK_ELEMS / 4 / 256, 256, 0, stream>>>(Q, K, Qh, Kh);
    // 1) scores via WMMA: 1024 strips, 4 waves/block
    dca_qk_wmma_kernel<<<Hn * (Ln / 16) / 4, 128, 0, stream>>>(Qh, Kh, Sbuf);
    // 2) row softmax in place: 16384 rows, 8 waves/block
    dca_softmax_kernel<<<Hn * Ln / 8, 256, 0, stream>>>(Sbuf);
    // 3) energy reduction
    hipMemsetAsync(out, 0, Bn * sizeof(float), stream);
    dca_energy_kernel<<<Bn * Hn, 256, 0, stream>>>(Sbuf, x, V, out);
}
